// NeuralImageStitching_6098853560994
// MI455X (gfx1250) — matrix operand order repaired and checked by reading, not compiled, not run on hardware
//
#include <hip/hip_runtime.h>

typedef __attribute__((ext_vector_type(16))) _Float16 v16h;
typedef __attribute__((ext_vector_type(8)))  _Float16 v8h;
typedef __attribute__((ext_vector_type(8)))  float    v8f;

#define HIDC 256
#define ENCC 64
#define HIN  256
#define WIN  256
#define HQ   384
#define WQ   384
#define NQ   (HQ*WQ)
#define NPIX (HIN*WIN)
#define PI_F 3.14159265358979323846f

union Frag16 { v16h h; uint4 u[2]; };
union Pack8  { v8h  h; uint4 u;    };

static __device__ __forceinline__ v8f vzero8() {
  v8f z = {0.f,0.f,0.f,0.f,0.f,0.f,0.f,0.f};
  return z;
}
static __device__ __forceinline__ v16h zero16h() {
  Frag16 f; f.u[0] = make_uint4(0,0,0,0); f.u[1] = make_uint4(0,0,0,0); return f.h;
}
// 16 contiguous halves (32B): B-fragment, K = base..base+15 for this lane
static __device__ __forceinline__ v16h load_run16(const _Float16* p) {
  Frag16 f;
  f.u[0] = *(const uint4*)(p);
  f.u[1] = *(const uint4*)(p + 8);
  return f.h;
}
// A-fragment: K run {kk*32+16*hi .. +15} (koff = kk*32 + 8*hi)
static __device__ __forceinline__ v16h load_afrag(const _Float16* wrow, int koff) {
  Frag16 f;
  f.u[0] = *(const uint4*)(wrow + koff + (koff & 31));
  f.u[1] = *(const uint4*)(wrow + koff + (koff & 31) + 8);
  return f.h;
}
static __device__ __forceinline__ v8f wmma16(v16h a, v16h b, v8f c) {
  return __builtin_amdgcn_wmma_f32_16x16x32_f16(false, a, false, b, (short)0, c, false, false);
}

// ---------------- weight repack: OIHW f32 -> [O][tap*I + i] fp16 ----------------
__global__ void k_repack_oihw(const float* __restrict__ w, _Float16* __restrict__ out,
                              int O, int I) {
  int n  = O * I * 9;
  int id = blockIdx.x * blockDim.x + threadIdx.x;
  if (id >= n) return;
  int o   = id / (I * 9);
  int rem = id - o * I * 9;
  int t   = rem / I;
  int i   = rem - t * I;
  out[o * I * 9 + t * I + i] = (_Float16)w[(o * I + i) * 9 + t];
}

__global__ void k_f32_to_f16(const float* __restrict__ in, _Float16* __restrict__ out, int n) {
  int id = blockIdx.x * blockDim.x + threadIdx.x;
  if (id < n) out[id] = (_Float16)in[id];
}

// ---------------- enc conv: 3 -> 64, 3x3 SAME @256x256, NCHW f32 -> HWC fp16 ----
__global__ __launch_bounds__(256) void k_enc_conv(const float* __restrict__ img,
                                                  const float* __restrict__ w,
                                                  const float* __restrict__ b,
                                                  _Float16* __restrict__ featT) {
  int id  = blockIdx.x * blockDim.x + threadIdx.x;   // NPIX*64 threads
  int oc  = id & 63;
  int pix = id >> 6;
  int y = pix >> 8, x = pix & 255;
  float acc = b[oc];
  #pragma unroll
  for (int c = 0; c < 3; c++) {
    #pragma unroll
    for (int ky = 0; ky < 3; ky++) {
      int sy = y + ky - 1;
      if (sy < 0 || sy >= HIN) continue;
      #pragma unroll
      for (int kx = 0; kx < 3; kx++) {
        int sx = x + kx - 1;
        if (sx < 0 || sx >= WIN) continue;
        acc += img[(c << 16) + (sy << 8) + sx] * w[oc * 27 + c * 9 + ky * 3 + kx];
      }
    }
  }
  featT[(size_t)pix * ENCC + oc] = (_Float16)acc;
}

// ---------------- coef/freq conv: 64 -> 256, 3x3 SAME @256x256, WMMA ------------
// one wave = 16 pixels x 256 out-ch ; K = 9 taps * 64 ch = 18 chunks of 32
__global__ __launch_bounds__(256) void k_conv_cf_wmma(const _Float16* __restrict__ featT,
                                                      const _Float16* __restrict__ wB,
                                                      const float* __restrict__ bias,
                                                      _Float16* __restrict__ outT) {
  int gw   = (blockIdx.x * blockDim.x + threadIdx.x) >> 5;   // 0..4095 pixel tiles
  int lane = threadIdx.x & 31;
  int m = lane & 15, hi = lane >> 4;
  int y  = gw >> 4;                 // 16 tiles per 256-px row
  int x0 = (gw & 15) << 4;
  int px = x0 + m;

  v8f acc[16];
  #pragma unroll
  for (int t = 0; t < 16; t++) acc[t] = vzero8();

  #pragma unroll
  for (int kk = 0; kk < 18; kk++) {
    int tap = kk >> 1;
    int ky = tap / 3, kx = tap - ky * 3;
    int sy = y + ky - 1;
    int sx = px + kx - 1;
    v16h bf;
    if (sy >= 0 && sy < HIN && sx >= 0 && sx < WIN)
      bf = load_run16(featT + ((size_t)((sy << 8) + sx)) * ENCC + ((kk & 1) << 5) + (hi << 4));
    else
      bf = zero16h();
    #pragma unroll
    for (int t = 0; t < 16; t++) {
      v16h af = load_afrag(wB + (t * 16 + m) * 576, kk * 32 + (hi << 3));
      acc[t] = wmma16(af, bf, acc[t]);
    }
  }

  int pix = (y << 8) + px;
  #pragma unroll
  for (int t = 0; t < 16; t++) {
    int fb = t * 16 + (hi << 3);
    Pack8 o;
    #pragma unroll
    for (int r = 0; r < 8; r++) o.h[r] = (_Float16)(acc[t][r] + bias[fb + r]);
    *(uint4*)(outT + (size_t)pix * HIDC + fb) = o.u;   // HWC fp16, 16B store
  }
}

// ---------------- neural warp: gather + fourier modulation ----------------------
__global__ __launch_bounds__(128) void k_warp(const _Float16* __restrict__ coefT,
                                              const _Float16* __restrict__ freqT,
                                              const float* __restrict__ grid,
                                              const float* __restrict__ cell,
                                              const float* __restrict__ mask,
                                              const float* __restrict__ phase_w,
                                              _Float16* __restrict__ fea,
                                              int chanOff) {
  int q = blockIdx.x;      // 0..NQ-1
  int k = threadIdx.x;     // 0..127
  float cy = grid[q * 2 + 0], cx = grid[q * 2 + 1];
  int iy = (int)rintf((cy + 1.f) * (HIN * 0.5f) - 0.5f);
  int ix = (int)rintf((cx + 1.f) * (WIN * 0.5f) - 0.5f);
  bool valid = (iy >= 0) & (iy < HIN) & (ix >= 0) & (ix < WIN);
  int iyc = min(max(iy, 0), HIN - 1), ixc = min(max(ix, 0), WIN - 1);
  int p = (iyc << 8) + ixc;
  float wcy = valid ? (-1.f + (2.f * iyc + 1.f) * (1.f / HIN)) : 0.f;
  float wcx = valid ? (-1.f + (2.f * ixc + 1.f) * (1.f / WIN)) : 0.f;
  float rely = (cy - wcy) * (float)HIN;
  float relx = (cx - wcx) * (float)WIN;
  float ph = 0.f;
  #pragma unroll
  for (int c = 0; c < 10; c++) ph += cell[q * 10 + c] * 256.f * phase_w[k * 10 + c];
  union { unsigned u; _Float16 h[2]; } fq;
  fq.u = *(const unsigned*)(freqT + (size_t)p * HIDC + 2 * k);
  float f0 = valid ? (float)fq.h[0] : 0.f;
  float f1 = valid ? (float)fq.h[1] : 0.f;
  float wf = f0 * rely + f1 * relx + ph;
  float c0 = valid ? (float)coefT[(size_t)p * HIDC + k] : 0.f;
  float c1 = valid ? (float)coefT[(size_t)p * HIDC + 128 + k] : 0.f;
  float mk = mask[q];
  float s, co;
  __sincosf(PI_F * wf, &s, &co);
  fea[(size_t)q * 512 + chanOff + k]       = (_Float16)(c0 * co * mk);
  fea[(size_t)q * 512 + chanOff + 128 + k] = (_Float16)(c1 * s * mk);
}

// ---------------- bld conv: 512 -> 64, 3x3 SAME @384x384, WMMA ------------------
// one wave = 32 pixels (2 N-tiles) x 64 out-ch ; K = 9*512 = 144 chunks
__global__ __launch_bounds__(256) void k_conv_bld_wmma(const _Float16* __restrict__ fea,
                                                       const _Float16* __restrict__ wB,
                                                       const float* __restrict__ bias,
                                                       _Float16* __restrict__ outT) {
  int wu   = (blockIdx.x * blockDim.x + threadIdx.x) >> 5;  // 0..4607
  int lane = threadIdx.x & 31;
  int m = lane & 15, hi = lane >> 4;
  int tile0 = wu * 2;                 // 24 tiles per 384-px row; pair stays in-row
  int y  = tile0 / 24;
  int x0 = (tile0 - y * 24) * 16;

  v8f acc0[4], acc1[4];
  #pragma unroll
  for (int t = 0; t < 4; t++) { acc0[t] = vzero8(); acc1[t] = vzero8(); }

  for (int kk = 0; kk < 144; kk++) {
    __builtin_prefetch(wB + m * 4608 + kk * 32 + 1024, 0, 1);
    int tap = kk >> 4;
    int ky = tap / 3, kx = tap - ky * 3;
    int sy = y + ky - 1;
    int cb = ((kk & 15) << 5) + (hi << 4);
    int sxa = x0 + m + kx - 1;
    int sxb = sxa + 16;               // always >= 15, only upper bound matters
    bool yok = (sy >= 0) & (sy < HQ);
    v16h bf0, bf1;
    if (yok && sxa >= 0 && sxa < WQ)
      bf0 = load_run16(fea + ((size_t)(sy * WQ + sxa)) * 512 + cb);
    else bf0 = zero16h();
    if (yok && sxb < WQ)
      bf1 = load_run16(fea + ((size_t)(sy * WQ + sxb)) * 512 + cb);
    else bf1 = zero16h();
    #pragma unroll
    for (int t = 0; t < 4; t++) {
      v16h af = load_afrag(wB + (t * 16 + m) * 4608, kk * 32 + (hi << 3));
      acc0[t] = wmma16(af, bf0, acc0[t]);
      acc1[t] = wmma16(af, bf1, acc1[t]);
    }
  }

  #pragma unroll
  for (int n = 0; n < 2; n++) {
    int pix = y * WQ + x0 + n * 16 + m;
    #pragma unroll
    for (int t = 0; t < 4; t++) {
      int fb = t * 16 + (hi << 3);
      v8f a = n ? acc1[t] : acc0[t];
      Pack8 o;
      #pragma unroll
      for (int r = 0; r < 8; r++) o.h[r] = (_Float16)(a[r] + bias[fb + r]);
      *(uint4*)(outT + (size_t)pix * 64 + fb) = o.u;
    }
  }
}

// ---------------- fused gather + MLP (WMMA through per-wave LDS) ----------------
template <int KCH>
static __device__ __forceinline__ void mlp_layer(const _Float16* __restrict__ W,
                                                 const float* __restrict__ bias,
                                                 const _Float16* xin, _Float16* xout,
                                                 int m, int hi, bool do_relu) {
  v8f acc[16];
  #pragma unroll
  for (int t = 0; t < 16; t++) acc[t] = vzero8();
  #pragma unroll
  for (int kk = 0; kk < KCH; kk++) {
    v16h bf = load_run16(xin + m * (KCH * 32) + kk * 32 + hi * 16);
    #pragma unroll
    for (int t = 0; t < 16; t++) {
      v16h af = load_afrag(W + (t * 16 + m) * (KCH * 32), kk * 32 + hi * 8);
      acc[t] = wmma16(af, bf, acc[t]);
    }
  }
  #pragma unroll
  for (int t = 0; t < 16; t++) {
    int fb = t * 16 + hi * 8;
    Pack8 o;
    #pragma unroll
    for (int r = 0; r < 8; r++) {
      float v = acc[t][r] + bias[fb + r];
      if (do_relu) v = fmaxf(v, 0.f);
      o.h[r] = (_Float16)v;
    }
    *(uint4*)(xout + m * 256 + fb) = o.u;
  }
}

__global__ __launch_bounds__(64) void k_mlp_wmma(const _Float16* __restrict__ bldo,
                                                 const float* __restrict__ coords,
                                                 const _Float16* __restrict__ w1h, const float* __restrict__ b1,
                                                 const _Float16* __restrict__ w2h, const float* __restrict__ b2,
                                                 const _Float16* __restrict__ w3h, const float* __restrict__ b3,
                                                 const float* __restrict__ w4, const float* __restrict__ b4,
                                                 float* __restrict__ out) {
  __shared__ __align__(16) _Float16 s0[2][16 * 64];
  __shared__ __align__(16) _Float16 sa[2][16 * 256];
  __shared__ __align__(16) _Float16 sb[2][16 * 256];
  int wv   = threadIdx.x >> 5;
  int lane = threadIdx.x & 31;
  int m = lane & 15, hi = lane >> 4;
  int qbase = (blockIdx.x * 2 + wv) * 16;

  // gather 64-ch bld features at nearest sample (zero if invalid)
  {
    int q = qbase + m;
    float cy = coords[q * 2 + 0], cx = coords[q * 2 + 1];
    int iy = (int)rintf((cy + 1.f) * (HQ * 0.5f) - 0.5f);
    int ix = (int)rintf((cx + 1.f) * (WQ * 0.5f) - 0.5f);
    bool valid = (iy >= 0) & (iy < HQ) & (ix >= 0) & (ix < WQ);
    int iyc = min(max(iy, 0), HQ - 1), ixc = min(max(ix, 0), WQ - 1);
    const uint4* src = (const uint4*)(bldo + ((size_t)(iyc * WQ + ixc)) * 64 + hi * 32);
    uint4 z = make_uint4(0, 0, 0, 0);
    uint4* dst = (uint4*)(&s0[wv][m * 64 + hi * 32]);
    dst[0] = valid ? src[0] : z;
    dst[1] = valid ? src[1] : z;
    dst[2] = valid ? src[2] : z;
    dst[3] = valid ? src[3] : z;
  }
  // LDS ops from one wave are in-order -> no barrier needed between stages
  mlp_layer<2>(w1h, b1, &s0[wv][0], &sa[wv][0], m, hi, true);
  mlp_layer<8>(w2h, b2, &sa[wv][0], &sb[wv][0], m, hi, true);
  mlp_layer<8>(w3h, b3, &sb[wv][0], &sa[wv][0], m, hi, true);

  if (hi == 0) {
    int q = qbase + m;
    float a0 = b4[0], a1 = b4[1], a2 = b4[2];
    for (int c = 0; c < 256; c++) {
      float xv = (float)sa[wv][m * 256 + c];
      a0 += xv * w4[c];
      a1 += xv * w4[256 + c];
      a2 += xv * w4[512 + c];
    }
    out[(size_t)q * 3 + 0] = a0;
    out[(size_t)q * 3 + 1] = a1;
    out[(size_t)q * 3 + 2] = a2;
  }
}

// ---------------------------------- launcher ------------------------------------
extern "C" void kernel_launch(void* const* d_in, const int* in_sizes, int n_in,
                              void* d_out, int out_size, void* d_ws, size_t ws_size,
                              hipStream_t stream) {
  const float* ref      = (const float*)d_in[0];
  const float* ref_grid = (const float*)d_in[1];
  const float* ref_cell = (const float*)d_in[2];
  const float* ref_mask = (const float*)d_in[3];
  const float* tgt      = (const float*)d_in[4];
  const float* tgt_grid = (const float*)d_in[5];
  const float* tgt_cell = (const float*)d_in[6];
  const float* tgt_mask = (const float*)d_in[7];
  const float* stit     = (const float*)d_in[8];
  // d_in[9], d_in[10] are h, w scalars (fixed 384x384; hardcoded)
  const float* enc_w   = (const float*)d_in[11];
  const float* enc_b   = (const float*)d_in[12];
  const float* coef_w  = (const float*)d_in[13];
  const float* coef_b  = (const float*)d_in[14];
  const float* freq_w  = (const float*)d_in[15];
  const float* freq_b  = (const float*)d_in[16];
  const float* phase_w = (const float*)d_in[17];
  const float* bld_w   = (const float*)d_in[18];
  const float* bld_b   = (const float*)d_in[19];
  const float* w1 = (const float*)d_in[20];
  const float* b1 = (const float*)d_in[21];
  const float* w2 = (const float*)d_in[22];
  const float* b2 = (const float*)d_in[23];
  const float* w3 = (const float*)d_in[24];
  const float* b3 = (const float*)d_in[25];
  const float* w4 = (const float*)d_in[26];
  const float* b4 = (const float*)d_in[27];
  float* out = (float*)d_out;
  (void)in_sizes; (void)n_in; (void)out_size; (void)ws_size;

  size_t off = 0;
  auto take = [&](size_t bytes) -> void* {
    void* p = (void*)((char*)d_ws + off);
    off += (bytes + 255) & ~(size_t)255;
    return p;
  };
  _Float16* wBc    = (_Float16*)take((size_t)256 * 576 * 2);
  _Float16* wBf    = (_Float16*)take((size_t)256 * 576 * 2);
  _Float16* wBb    = (_Float16*)take((size_t)64 * 4608 * 2);
  _Float16* w1h    = (_Float16*)take((size_t)256 * 64 * 2);
  _Float16* w2h    = (_Float16*)take((size_t)256 * 256 * 2);
  _Float16* w3h    = (_Float16*)take((size_t)256 * 256 * 2);
  _Float16* featR  = (_Float16*)take((size_t)NPIX * 64 * 2);
  _Float16* featTg = (_Float16*)take((size_t)NPIX * 64 * 2);
  _Float16* coefR  = (_Float16*)take((size_t)NPIX * 256 * 2);
  _Float16* freqR  = (_Float16*)take((size_t)NPIX * 256 * 2);
  _Float16* coefT  = (_Float16*)take((size_t)NPIX * 256 * 2);
  _Float16* freqT  = (_Float16*)take((size_t)NPIX * 256 * 2);
  _Float16* fea    = (_Float16*)take((size_t)NQ * 512 * 2);
  _Float16* bldo   = (_Float16*)take((size_t)NQ * 64 * 2);

  { int n = 256 * 64 * 9; k_repack_oihw<<<(n + 255) / 256, 256, 0, stream>>>(coef_w, wBc, 256, 64); }
  { int n = 256 * 64 * 9; k_repack_oihw<<<(n + 255) / 256, 256, 0, stream>>>(freq_w, wBf, 256, 64); }
  { int n = 64 * 512 * 9; k_repack_oihw<<<(n + 255) / 256, 256, 0, stream>>>(bld_w, wBb, 64, 512); }
  { int n = 256 * 64;  k_f32_to_f16<<<(n + 255) / 256, 256, 0, stream>>>(w1, w1h, n); }
  { int n = 256 * 256; k_f32_to_f16<<<(n + 255) / 256, 256, 0, stream>>>(w2, w2h, n); }
  { int n = 256 * 256; k_f32_to_f16<<<(n + 255) / 256, 256, 0, stream>>>(w3, w3h, n); }

  k_enc_conv<<<(NPIX * 64) / 256, 256, 0, stream>>>(ref, enc_w, enc_b, featR);
  k_enc_conv<<<(NPIX * 64) / 256, 256, 0, stream>>>(tgt, enc_w, enc_b, featTg);

  k_conv_cf_wmma<<<512, 256, 0, stream>>>(featR,  wBc, coef_b, coefR);
  k_conv_cf_wmma<<<512, 256, 0, stream>>>(featR,  wBf, freq_b, freqR);
  k_conv_cf_wmma<<<512, 256, 0, stream>>>(featTg, wBc, coef_b, coefT);
  k_conv_cf_wmma<<<512, 256, 0, stream>>>(featTg, wBf, freq_b, freqT);

  k_warp<<<NQ, 128, 0, stream>>>(coefR, freqR, ref_grid, ref_cell, ref_mask, phase_w, fea, 0);
  k_warp<<<NQ, 128, 0, stream>>>(coefT, freqT, tgt_grid, tgt_cell, tgt_mask, phase_w, fea, 256);

  k_conv_bld_wmma<<<576, 256, 0, stream>>>(fea, wBb, bld_b, bldo);

  k_mlp_wmma<<<4608, 64, 0, stream>>>(bldo, stit, w1h, b1, w2h, b2, w3h, b3, w4, b4, out);
}
